// TextDecoder_64390149702371
// MI455X (gfx1250) — matrix operand order repaired and checked by reading, not compiled, not run on hardware
//
#include <hip/hip_runtime.h>
#include <hip/hip_bf16.h>

// ---------------- problem constants ----------------
#define BSZ    16
#define SEQ    256
#define NVIS   196
#define DMODEL 768
#define FFDIM  2048
#define VOCAB  30522
#define NLAYER 6
#define NHEAD  8
#define HD     96   // DMODEL / NHEAD

typedef __attribute__((ext_vector_type(16))) __bf16 v16bf;
typedef __attribute__((ext_vector_type(8)))  float  v8f;
typedef __attribute__((ext_vector_type(2)))  __bf16 bf16x2;

// ---------- optional CDNA5 async global->LDS path (probe-guarded) ----------
#if defined(__has_builtin)
#  if __has_builtin(__builtin_amdgcn_global_load_async_to_lds_b64)
#    define USE_ASYNC_LDS 1
#  endif
#  if __has_builtin(__builtin_amdgcn_s_wait_asynccnt)
#    define HAVE_WAIT_ASYNC 1
#  endif
#endif

#ifdef USE_ASYNC_LDS
// Builtin signature (from hipcc diagnostic): param0 is <2 x i32> addrspace(1)*,
// i.e. payload-typed pointers. b64 payload = v2i.
typedef int v2i __attribute__((ext_vector_type(2)));
typedef __attribute__((address_space(1))) v2i* gas1_p;
typedef __attribute__((address_space(3))) v2i* las3_p;
static __device__ __forceinline__ void async_copy8(const void* g, void* l) {
    // ISA: GLOBAL_LOAD_ASYNC_TO_LDS_B64 (ASYNCcnt). Generic->AS1 via integer
    // round-trip; generic LDS addr low 32 bits are the LDS offset (ISA 10.2).
    __builtin_amdgcn_global_load_async_to_lds_b64(
        (gas1_p)(uintptr_t)g,
        (las3_p)(unsigned int)(uintptr_t)l, 0, 0);
}
#endif

static __device__ __forceinline__ void async_drain() {
#ifdef HAVE_WAIT_ASYNC
    __builtin_amdgcn_s_wait_asynccnt(0);
#endif
}

// ---------------- elementwise helpers ----------------
__global__ void k_f32_to_bf16(const float* __restrict__ s, __bf16* __restrict__ d, long n) {
    long i = (long)blockIdx.x * blockDim.x + threadIdx.x;
    long st = (long)gridDim.x * blockDim.x;
    for (; i < n; i += st) d[i] = (__bf16)s[i];
}

// x = target_embed + sinusoid PE ; also emit bf16 copy
__global__ void k_add_pe(const float* __restrict__ te, float* __restrict__ x,
                         __bf16* __restrict__ xb, long n) {
    long i = (long)blockIdx.x * blockDim.x + threadIdx.x;
    long st = (long)gridDim.x * blockDim.x;
    const float c = -logf(10000.0f) / (float)DMODEL;
    for (; i < n; i += st) {
        int d = (int)(i % DMODEL);
        int s = (int)((i / DMODEL) % SEQ);
        float freq = __expf(c * (float)(d & ~1));
        float ang = (float)s * freq;
        float pe = (d & 1) ? __cosf(ang) : __sinf(ang);
        float v = te[i] + pe;
        x[i] = v;
        xb[i] = (__bf16)v;
    }
}

// fused QKV split: (B,SEQ,3D) f32 -> q (B,NH,SEQ,HD), k (B,NH,SEQ,HD), vT (B,NH,HD,SEQ), all bf16
__global__ void k_split_qkv(const float* __restrict__ src, __bf16* __restrict__ q,
                            __bf16* __restrict__ k, __bf16* __restrict__ vT, long n) {
    long i = (long)blockIdx.x * blockDim.x + threadIdx.x;
    long st = (long)gridDim.x * blockDim.x;
    for (; i < n; i += st) {
        int col = (int)(i % (3 * DMODEL));
        long bs = i / (3 * DMODEL);
        int s = (int)(bs % SEQ);
        int b = (int)(bs / SEQ);
        int part = col / DMODEL, d = col % DMODEL;
        int hh = d / HD, dd = d % HD;
        __bf16 v = (__bf16)src[i];
        long bh = (long)b * NHEAD + hh;
        if (part == 0)      q [(bh * SEQ + s) * HD + dd] = v;
        else if (part == 1) k [(bh * SEQ + s) * HD + dd] = v;
        else                vT[(bh * HD + dd) * SEQ + s] = v;
    }
}

// q-only split: (B,SEQ,D) f32 -> (B,NH,SEQ,HD) bf16
__global__ void k_split_q(const float* __restrict__ src, __bf16* __restrict__ q, long n) {
    long i = (long)blockIdx.x * blockDim.x + threadIdx.x;
    long st = (long)gridDim.x * blockDim.x;
    for (; i < n; i += st) {
        int d = (int)(i % DMODEL);
        long bs = i / DMODEL;
        int s = (int)(bs % SEQ);
        int b = (int)(bs / SEQ);
        int hh = d / HD, dd = d % HD;
        q[(((long)b * NHEAD + hh) * SEQ + s) * HD + dd] = (__bf16)src[i];
    }
}

// fused KV split for cross-attn: (B,NVIS,2D) f32 -> k (B,NH,NVIS,HD), vT (B,NH,HD,NVIS)
__global__ void k_split_kv(const float* __restrict__ src, __bf16* __restrict__ k,
                           __bf16* __restrict__ vT, long n) {
    long i = (long)blockIdx.x * blockDim.x + threadIdx.x;
    long st = (long)gridDim.x * blockDim.x;
    for (; i < n; i += st) {
        int col = (int)(i % (2 * DMODEL));
        long bs = i / (2 * DMODEL);
        int s = (int)(bs % NVIS);
        int b = (int)(bs / NVIS);
        int part = col / DMODEL, d = col % DMODEL;
        int hh = d / HD, dd = d % HD;
        __bf16 v = (__bf16)src[i];
        long bh = (long)b * NHEAD + hh;
        if (part == 0) k [(bh * NVIS + s) * HD + dd] = v;
        else           vT[(bh * HD + dd) * NVIS + s] = v;
    }
}

// (B, NH, SEQ, HD) f32 -> (B, SEQ, D) bf16
__global__ void k_merge_heads(const float* __restrict__ src, __bf16* __restrict__ dst, long n) {
    long i = (long)blockIdx.x * blockDim.x + threadIdx.x;
    long st = (long)gridDim.x * blockDim.x;
    for (; i < n; i += st) {
        int d = (int)(i % DMODEL);
        long bs = i / DMODEL;
        int s = (int)(bs % SEQ);
        int b = (int)(bs / SEQ);
        int hh = d / HD, dd = d % HD;
        dst[i] = (__bf16)src[(((long)b * NHEAD + hh) * SEQ + s) * HD + dd];
    }
}

// gather dec = x_bf16[:, :S-1, :]
__global__ void k_dec_gather(const __bf16* __restrict__ xb, __bf16* __restrict__ dec, long n) {
    long i = (long)blockIdx.x * blockDim.x + threadIdx.x;
    long st = (long)gridDim.x * blockDim.x;
    for (; i < n; i += st) {
        int d = (int)(i % DMODEL);
        long bs = i / DMODEL;
        int s = (int)(bs % (SEQ - 1));
        int b = (int)(bs / (SEQ - 1));
        dec[i] = xb[((long)b * SEQ + s) * DMODEL + d];
    }
}

// ---------------- softmax over attention rows (Lk <= 256) ----------------
__global__ __launch_bounds__(256)
void k_attn_softmax(const float* __restrict__ sc, __bf16* __restrict__ at,
                    int Lq, int Lk, int causal) {
    __shared__ float red[256];
    long row = blockIdx.x;                      // (b*NH + h)*Lq + q
    int q = (int)(row % Lq);
    const float* srow = sc + row * (long)Lk;
    __bf16* arow = at + row * (long)Lk;
    int tid = threadIdx.x;
    float v = -INFINITY;
    if (tid < Lk) {
        v = srow[tid];
        if (causal && tid > q) v = -1e9f;
    }
    red[tid] = v; __syncthreads();
    for (int o = 128; o > 0; o >>= 1) { if (tid < o) red[tid] = fmaxf(red[tid], red[tid + o]); __syncthreads(); }
    float m = red[0]; __syncthreads();
    float e = (tid < Lk) ? __expf(v - m) : 0.0f;
    red[tid] = e; __syncthreads();
    for (int o = 128; o > 0; o >>= 1) { if (tid < o) red[tid] += red[tid + o]; __syncthreads(); }
    float s = red[0];
    if (tid < Lk) arow[tid] = (__bf16)(e / s);
}

// ---------------- residual + LayerNorm over D=768 (256 thr x 3) ----------------
__global__ __launch_bounds__(256)
void k_residual_ln(const float* __restrict__ res, const float* __restrict__ br,
                   const float* __restrict__ w, const float* __restrict__ bia,
                   float* __restrict__ xo, __bf16* __restrict__ xbo) {
    __shared__ float red[256];
    long row = blockIdx.x;
    const float* r0 = res + row * (long)DMODEL;
    const float* b0 = br + row * (long)DMODEL;
    int tid = threadIdx.x;
    float v[3]; float s = 0.0f;
    #pragma unroll
    for (int j = 0; j < 3; ++j) { int i = tid + 256 * j; v[j] = r0[i] + b0[i]; s += v[j]; }
    red[tid] = s; __syncthreads();
    for (int o = 128; o > 0; o >>= 1) { if (tid < o) red[tid] += red[tid + o]; __syncthreads(); }
    float mean = red[0] * (1.0f / DMODEL); __syncthreads();
    float s2 = 0.0f;
    #pragma unroll
    for (int j = 0; j < 3; ++j) { float d = v[j] - mean; s2 += d * d; }
    red[tid] = s2; __syncthreads();
    for (int o = 128; o > 0; o >>= 1) { if (tid < o) red[tid] += red[tid + o]; __syncthreads(); }
    float inv = rsqrtf(red[0] * (1.0f / DMODEL) + 1e-5f);
    #pragma unroll
    for (int j = 0; j < 3; ++j) {
        int i = tid + 256 * j;
        float o2 = (v[j] - mean) * inv * w[i] + bia[i];
        xo[row * (long)DMODEL + i] = o2;
        xbo[row * (long)DMODEL + i] = (__bf16)o2;
    }
}

// ---------------- F_t = softmax(dec, axis=-1) over D ----------------
__global__ __launch_bounds__(256)
void k_ft_softmax(const float* __restrict__ x, float* __restrict__ out) {
    __shared__ float red[256];
    long row = blockIdx.x;                    // b*(S-1)+s
    int s = (int)(row % (SEQ - 1));
    int b = (int)(row / (SEQ - 1));
    const float* xr = x + ((long)b * SEQ + s) * DMODEL;
    int tid = threadIdx.x;
    float v[3]; float m = -INFINITY;
    #pragma unroll
    for (int j = 0; j < 3; ++j) { v[j] = xr[tid + 256 * j]; m = fmaxf(m, v[j]); }
    red[tid] = m; __syncthreads();
    for (int o = 128; o > 0; o >>= 1) { if (tid < o) red[tid] = fmaxf(red[tid], red[tid + o]); __syncthreads(); }
    m = red[0]; __syncthreads();
    float e[3]; float sum = 0.0f;
    #pragma unroll
    for (int j = 0; j < 3; ++j) { e[j] = __expf(v[j] - m); sum += e[j]; }
    red[tid] = sum; __syncthreads();
    for (int o = 128; o > 0; o >>= 1) { if (tid < o) red[tid] += red[tid + o]; __syncthreads(); }
    float inv = 1.0f / red[0];
    #pragma unroll
    for (int j = 0; j < 3; ++j) out[row * (long)DMODEL + tid + 256 * j] = e[j] * inv;
}

// ---------------- batched bf16 WMMA GEMM (NT only) ----------------
// C[b] = act(alpha * A[b] @ B[b]^T + bias), A:(M,K) row-major, B:(N,K) row-major.
// 64x64 block tile, 4 waves, 2x2 v_wmma_f32_16x16x32_bf16 per wave, BK=32.
// Fast path: 8-byte async->LDS staging (ASYNCcnt) with sync fallback; edge
// path: clamped unconditional loads + select (no exec branching).
__global__ __launch_bounds__(128)
void k_gemm_wmma(const __bf16* __restrict__ A, long lda, long strideA,
                 const __bf16* __restrict__ Bm, long ldb, long strideB,
                 const float* __restrict__ bias,
                 float* __restrict__ C, long ldc, long strideC,
                 int M, int N, int K, float alpha, int relu) {
    __shared__ __bf16 As[64][36];   // 72B row stride: 8B-aligned vector stores, low conflict
    __shared__ __bf16 Bs[64][36];   // n-major [n][k]

    const int bz = blockIdx.z;
    A  += (long)bz * strideA;
    Bm += (long)bz * strideB;
    C  += (long)bz * strideC;

    const int rowBase = blockIdx.y * 64;
    const int colBase = blockIdx.x * 64;
    const int t = threadIdx.x;
    const int wave = t >> 5, lane = t & 31;
    const int wm = wave >> 1, wn = wave & 1;
    const int m16 = lane & 15, h = lane >> 4;

    const bool fullM = (rowBase + 64 <= M);
    const bool fullN = (colBase + 64 <= N);

    v8f acc[2][2] = {};

    for (int k0 = 0; k0 < K; k0 += 32) {
        const bool fullK = (k0 + 32 <= K);

        // ---- stage A tile (64 rows x 32 k) ----
        if (fullM && fullK) {
            for (int c = t; c < 256; c += 128) {          // 8 bytes (4 bf16) copied per chunk
                int r = c >> 2, kk = (c & 3) * 8;
                const void* g = (const void*)(A + (long)(rowBase + r) * lda + (k0 + kk));
#ifdef USE_ASYNC_LDS
                async_copy8(g, &As[r][kk]);
#else
                *(unsigned long long*)&As[r][kk] = *(const unsigned long long*)g;
#endif
            }
        } else {
            for (int e = 0; e < 16; ++e) {
                int idx = t * 16 + e;
                int r = idx >> 5, kk = idx & 31;
                int gr = rowBase + r, gk = k0 + kk;
                bool ok = (gr < M) && (gk < K);
                long ia = (long)(gr < M ? gr : M - 1) * lda + (gk < K ? gk : K - 1);
                __bf16 v = A[ia];
                As[r][kk] = ok ? v : (__bf16)0.0f;
            }
        }
        // ---- stage B tile (64 n x 32 k) ----
        if (fullN && fullK) {
            for (int c = t; c < 256; c += 128) {
                int nn = c >> 2, kk = (c & 3) * 8;
                const void* g = (const void*)(Bm + (long)(colBase + nn) * ldb + (k0 + kk));
#ifdef USE_ASYNC_LDS
                async_copy8(g, &Bs[nn][kk]);
#else
                *(unsigned long long*)&Bs[nn][kk] = *(const unsigned long long*)g;
#endif
            }
        } else {
            for (int e = 0; e < 16; ++e) {
                int idx = t * 16 + e;
                int nn = idx >> 5, kk = idx & 31;
                int gn = colBase + nn, gk = k0 + kk;
                bool ok = (gn < N) && (gk < K);
                long ib = (long)(gn < N ? gn : N - 1) * ldb + (gk < K ? gk : K - 1);
                __bf16 v = Bm[ib];
                Bs[nn][kk] = ok ? v : (__bf16)0.0f;
            }
        }

        // prefetch next K tile into caches (global_prefetch_b8)
        if (k0 + 32 < K && lane == 0) {
            int pr = rowBase + wave * 16; if (pr >= M) pr = M - 1;
            __builtin_prefetch(&A[(long)pr * lda + k0 + 32], 0, 0);
            int pn = colBase + wave * 16; if (pn >= N) pn = N - 1;
            __builtin_prefetch(&Bm[(long)pn * ldb + k0 + 32], 0, 0);
        }

        async_drain();          // s_wait_asynccnt 0 (no-op when async path unused)
        __syncthreads();

        // ---- fragments per ISA 7.12.2 bf16 layouts ----
        v16bf afrag[2], bfrag[2];
        #pragma unroll
        for (int tm = 0; tm < 2; ++tm) {
            int mrow = wm * 32 + tm * 16 + m16;
            #pragma unroll
            for (int j = 0; j < 8; ++j) {
                int kk = ((j >= 4) ? 16 : 0) + (j & 3) * 2 + 8 * h;    // A 16x32 bf16
                bf16x2 p = *(const bf16x2*)&As[mrow][kk];
                afrag[tm][2 * j] = p.x; afrag[tm][2 * j + 1] = p.y;
            }
        }
        #pragma unroll
        for (int tn = 0; tn < 2; ++tn) {
            int nrow = wn * 32 + tn * 16 + m16;
            #pragma unroll
            for (int j = 0; j < 8; ++j) {
                int kk = ((j >= 4) ? 16 : 0) + (j & 3) * 2 + 8 * h;    // B 32x16 bf16
                bf16x2 p = *(const bf16x2*)&Bs[nrow][kk];
                bfrag[tn][2 * j] = p.x; bfrag[tn][2 * j + 1] = p.y;
            }
        }

        #pragma unroll
        for (int tm = 0; tm < 2; ++tm)
            #pragma unroll
            for (int tn = 0; tn < 2; ++tn)
                acc[tm][tn] = __builtin_amdgcn_wmma_f32_16x16x32_bf16(
                    false, afrag[tm], false, bfrag[tn],
                    (short)0, acc[tm][tn], false, false);
        __syncthreads();
    }

    // ---- store: C/D layout VGPR r -> M = r + 8*h, N = lane&15 ----
    #pragma unroll
    for (int tm = 0; tm < 2; ++tm) {
        #pragma unroll
        for (int tn = 0; tn < 2; ++tn) {
            int col = colBase + wn * 32 + tn * 16 + m16;
            if (col >= N) continue;
            float bv = bias ? bias[col] : 0.0f;
            #pragma unroll
            for (int r = 0; r < 8; ++r) {
                int row = rowBase + wm * 32 + tm * 16 + r + 8 * h;
                if (row < M) {
                    float v = alpha * acc[tm][tn][r] + bv;
                    if (relu) v = fmaxf(v, 0.0f);
                    C[(long)row * ldc + col] = v;
                }
            }
        }
    }
}

// ---------------- host-side helpers ----------------
static void conv(const float* s, __bf16* d, long n, hipStream_t st) {
    int blocks = (int)((n + 255) / 256);
    k_f32_to_bf16<<<blocks, 256, 0, st>>>(s, d, n);
}

static void gemm(const __bf16* A, long lda, long sA,
                 const __bf16* Bm, long ldb, long sB,
                 const float* bias, float* C, long ldc, long sC,
                 int M, int N, int K, int batch, float alpha, int relu, hipStream_t st) {
    dim3 g((N + 63) / 64, (M + 63) / 64, batch), b(128);
    k_gemm_wmma<<<g, b, 0, st>>>(A, lda, sA, Bm, ldb, sB, bias, C, ldc, sC, M, N, K, alpha, relu);
}

extern "C" void kernel_launch(void* const* d_in, const int* in_sizes, int n_in,
                              void* d_out, int out_size, void* d_ws, size_t ws_size,
                              hipStream_t stream) {
    (void)in_sizes; (void)n_in; (void)out_size; (void)ws_size;
    const float* fv          = (const float*)d_in[0];
    const float* te          = (const float*)d_in[1];
    const float* self_in_w   = (const float*)d_in[2];
    const float* self_in_b   = (const float*)d_in[3];
    const float* self_out_w  = (const float*)d_in[4];
    const float* self_out_b  = (const float*)d_in[5];
    const float* cross_in_w  = (const float*)d_in[6];
    const float* cross_in_b  = (const float*)d_in[7];
    const float* cross_out_w = (const float*)d_in[8];
    const float* cross_out_b = (const float*)d_in[9];
    const float* lin1_w      = (const float*)d_in[10];
    const float* lin1_b      = (const float*)d_in[11];
    const float* lin2_w      = (const float*)d_in[12];
    const float* lin2_b      = (const float*)d_in[13];
    const float* ln_w        = (const float*)d_in[14];
    const float* ln_b        = (const float*)d_in[15];
    const float* fc_w        = (const float*)d_in[16];

    // ---- carve workspace ----
    char* ws = (char*)d_ws;
    size_t off = 0;
    auto carve = [&](size_t bytes) -> char* {
        char* p = ws + off;
        off += (bytes + 255) & ~(size_t)255;
        return p;
    };
    const size_t D2 = (size_t)DMODEL * DMODEL;
    __bf16* w_self_in   = (__bf16*)carve((size_t)NLAYER * 3 * D2 * 2);
    __bf16* w_self_out  = (__bf16*)carve((size_t)NLAYER * D2 * 2);
    __bf16* w_cross_in  = (__bf16*)carve((size_t)NLAYER * 3 * D2 * 2);
    __bf16* w_cross_out = (__bf16*)carve((size_t)NLAYER * D2 * 2);
    __bf16* w_lin1      = (__bf16*)carve((size_t)NLAYER * FFDIM * DMODEL * 2);
    __bf16* w_lin2      = (__bf16*)carve((size_t)NLAYER * FFDIM * DMODEL * 2);
    __bf16* w_fc        = (__bf16*)carve((size_t)VOCAB * DMODEL * 2);
    float*  x_f   = (float*) carve((size_t)BSZ * SEQ * DMODEL * 4);
    __bf16* x_b   = (__bf16*)carve((size_t)BSZ * SEQ * DMODEL * 2);
    __bf16* mem_b = (__bf16*)carve((size_t)BSZ * NVIS * DMODEL * 2);
    float*  tmp0  = (float*) carve((size_t)BSZ * SEQ * 3 * DMODEL * 4); // QKV / FFN hidden
    float*  tmp1  = (float*) carve((size_t)BSZ * SEQ * DMODEL * 4);     // branch outputs
    __bf16* qb    = (__bf16*)carve((size_t)BSZ * NHEAD * SEQ * HD * 2);
    __bf16* kb    = (__bf16*)carve((size_t)BSZ * NHEAD * SEQ * HD * 2);
    __bf16* vT    = (__bf16*)carve((size_t)BSZ * NHEAD * HD * SEQ * 2); // transposed V
    float*  scor  = (float*) carve((size_t)BSZ * NHEAD * SEQ * SEQ * 4);
    __bf16* attnb = (__bf16*)carve((size_t)BSZ * NHEAD * SEQ * SEQ * 2);
    float*  o_f   = (float*) carve((size_t)BSZ * NHEAD * SEQ * HD * 4);
    __bf16* o_b   = (__bf16*)carve((size_t)BSZ * SEQ * DMODEL * 2);
    __bf16* h_b   = (__bf16*)carve((size_t)BSZ * SEQ * FFDIM * 2);
    __bf16* dec_b = (__bf16*)carve((size_t)BSZ * (SEQ - 1) * DMODEL * 2);

    const int  BS = BSZ * SEQ;    // 4096 query rows
    const int  BM = BSZ * NVIS;   // 3136 memory rows
    const long nXD = (long)BS * DMODEL;
    const float inv_sqrt_hd = 0.1020620726159658f;  // 1/sqrt(96)

    // ---- weight conversion fp32 -> bf16 (every call; no caching) ----
    conv(self_in_w,   w_self_in,   (long)NLAYER * 3 * D2, stream);
    conv(self_out_w,  w_self_out,  (long)NLAYER * D2, stream);
    conv(cross_in_w,  w_cross_in,  (long)NLAYER * 3 * D2, stream);
    conv(cross_out_w, w_cross_out, (long)NLAYER * D2, stream);
    conv(lin1_w,      w_lin1,      (long)NLAYER * FFDIM * DMODEL, stream);
    conv(lin2_w,      w_lin2,      (long)NLAYER * FFDIM * DMODEL, stream);
    conv(fc_w,        w_fc,        (long)VOCAB * DMODEL, stream);
    conv(fv,          mem_b,       (long)BM * DMODEL, stream);

    // ---- x = target_embed + PE ----
    k_add_pe<<<(int)((nXD + 255) / 256), 256, 0, stream>>>(te, x_f, x_b, nXD);

    for (int l = 0; l < NLAYER; ++l) {
        // ================= self attention (causal) =================
        {
            // fused QKV projection: (BS, D) @ (3D, D)^T -> (BS, 3D)
            gemm(x_b, DMODEL, 0, w_self_in + (size_t)l * 3 * D2, DMODEL, 0,
                 self_in_b + (size_t)l * 3 * DMODEL, tmp0, 3 * DMODEL, 0,
                 BS, 3 * DMODEL, DMODEL, 1, 1.0f, 0, stream);
            long nQKV = (long)BS * 3 * DMODEL;
            k_split_qkv<<<(int)((nQKV + 255) / 256), 256, 0, stream>>>(tmp0, qb, kb, vT, nQKV);

            // scores = q @ k^T / sqrt(hd), batched over B*NH
            gemm(qb, HD, (long)SEQ * HD, kb, HD, (long)SEQ * HD, nullptr,
                 scor, SEQ, (long)SEQ * SEQ, SEQ, SEQ, HD, BSZ * NHEAD, inv_sqrt_hd, 0, stream);
            k_attn_softmax<<<BSZ * NHEAD * SEQ, 256, 0, stream>>>(scor, attnb, SEQ, SEQ, 1);
            // o = attn @ vT^T
            gemm(attnb, SEQ, (long)SEQ * SEQ, vT, SEQ, (long)HD * SEQ, nullptr,
                 o_f, HD, (long)SEQ * HD, SEQ, HD, SEQ, BSZ * NHEAD, 1.0f, 0, stream);
            k_merge_heads<<<(int)((nXD + 255) / 256), 256, 0, stream>>>(o_f, o_b, nXD);
            gemm(o_b, DMODEL, 0, w_self_out + (size_t)l * D2, DMODEL, 0,
                 self_out_b + (size_t)l * DMODEL, tmp1, DMODEL, 0, BS, DMODEL, DMODEL, 1, 1.0f, 0, stream);
            k_residual_ln<<<BS, 256, 0, stream>>>(x_f, tmp1,
                 ln_w + ((size_t)l * 3 + 0) * DMODEL, ln_b + ((size_t)l * 3 + 0) * DMODEL, x_f, x_b);
        }
        // ================= cross attention =================
        {
            const __bf16* wq = w_cross_in + (size_t)l * 3 * D2;          // rows [0, D)
            const __bf16* wkv = wq + D2;                                  // rows [D, 3D)
            const float* bq = cross_in_b + (size_t)l * 3 * DMODEL;
            const float* bkv = bq + DMODEL;

            gemm(x_b, DMODEL, 0, wq, DMODEL, 0, bq, tmp0, DMODEL, 0,
                 BS, DMODEL, DMODEL, 1, 1.0f, 0, stream);
            k_split_q<<<(int)((nXD + 255) / 256), 256, 0, stream>>>(tmp0, qb, nXD);

            // fused K,V projection of memory: (BM, D) @ (2D, D)^T -> (BM, 2D)
            gemm(mem_b, DMODEL, 0, wkv, DMODEL, 0, bkv, tmp0, 2 * DMODEL, 0,
                 BM, 2 * DMODEL, DMODEL, 1, 1.0f, 0, stream);
            long nKV = (long)BM * 2 * DMODEL;
            k_split_kv<<<(int)((nKV + 255) / 256), 256, 0, stream>>>(tmp0, kb, vT, nKV);

            gemm(qb, HD, (long)SEQ * HD, kb, HD, (long)NVIS * HD, nullptr,
                 scor, NVIS, (long)SEQ * NVIS, SEQ, NVIS, HD, BSZ * NHEAD, inv_sqrt_hd, 0, stream);
            k_attn_softmax<<<BSZ * NHEAD * SEQ, 256, 0, stream>>>(scor, attnb, SEQ, NVIS, 0);
            gemm(attnb, NVIS, (long)SEQ * NVIS, vT, NVIS, (long)HD * NVIS, nullptr,
                 o_f, HD, (long)SEQ * HD, SEQ, HD, NVIS, BSZ * NHEAD, 1.0f, 0, stream);
            k_merge_heads<<<(int)((nXD + 255) / 256), 256, 0, stream>>>(o_f, o_b, nXD);
            gemm(o_b, DMODEL, 0, w_cross_out + (size_t)l * D2, DMODEL, 0,
                 cross_out_b + (size_t)l * DMODEL, tmp1, DMODEL, 0, BS, DMODEL, DMODEL, 1, 1.0f, 0, stream);
            k_residual_ln<<<BS, 256, 0, stream>>>(x_f, tmp1,
                 ln_w + ((size_t)l * 3 + 1) * DMODEL, ln_b + ((size_t)l * 3 + 1) * DMODEL, x_f, x_b);
        }
        // ================= FFN =================
        {
            gemm(x_b, DMODEL, 0, w_lin1 + (size_t)l * FFDIM * DMODEL, DMODEL, 0,
                 lin1_b + (size_t)l * FFDIM, tmp0, FFDIM, 0, BS, FFDIM, DMODEL, 1, 1.0f, 1, stream);
            conv(tmp0, h_b, (long)BS * FFDIM, stream);
            gemm(h_b, FFDIM, 0, w_lin2 + (size_t)l * FFDIM * DMODEL, FFDIM, 0,
                 lin2_b + (size_t)l * DMODEL, tmp1, DMODEL, 0, BS, DMODEL, FFDIM, 1, 1.0f, 0, stream);
            k_residual_ln<<<BS, 256, 0, stream>>>(x_f, tmp1,
                 ln_w + ((size_t)l * 3 + 2) * DMODEL, ln_b + ((size_t)l * 3 + 2) * DMODEL, x_f, x_b);
        }
    }

    // ---- outputs: logits (B,S-1,V) then F_t (B,S-1,D) ----
    const long nDec = (long)BSZ * (SEQ - 1) * DMODEL;
    k_dec_gather<<<(int)((nDec + 255) / 256), 256, 0, stream>>>(x_b, dec_b, nDec);

    float* out_logits = (float*)d_out;
    float* out_ft = out_logits + (size_t)BSZ * (SEQ - 1) * VOCAB;

    gemm(dec_b, DMODEL, 0, w_fc, DMODEL, 0, nullptr,
         out_logits, VOCAB, 0, BSZ * (SEQ - 1), VOCAB, DMODEL, 1, 1.0f, 0, stream);
    k_ft_softmax<<<BSZ * (SEQ - 1), 256, 0, stream>>>(x_f, out_ft);
}
